// NLBlockND_29429115912491
// MI455X (gfx1250) — matrix operand order repaired and checked by reading, not compiled, not run on hardware
//
#include <hip/hip_runtime.h>
#include <hip/hip_bf16.h>
#include <math.h>

// ---------------------------------------------------------------------------
// Types for CDNA5 WMMA (wave32): v_wmma_f32_16x16x32_bf16
// ---------------------------------------------------------------------------
typedef __attribute__((ext_vector_type(16))) __bf16 v16bf;
typedef __attribute__((ext_vector_type(8)))  __bf16 v8bf;
typedef __attribute__((ext_vector_type(8)))  float  v8f;

#define B_   128
#define C_   1024
#define HW_  192
#define CI_  512

static __device__ inline v8f vzero8() {
  v8f z;
#pragma unroll
  for (int i = 0; i < 8; ++i) z[i] = 0.0f;
  return z;
}

static __device__ inline v16bf cat16(v8bf lo, v8bf hi) {
  return __builtin_shufflevector(lo, hi, 0, 1, 2, 3, 4, 5, 6, 7,
                                          8, 9, 10, 11, 12, 13, 14, 15);
}

// CDNA5 async copy: 16 bytes global -> LDS per lane, tracked by ASYNCcnt.
// VDST carries the wave-relative LDS byte address (low 32 bits of a generic
// pointer to __shared__), HW adds LDS_BASE.
static __device__ inline void async_copy16(unsigned lds_dst, const void* gsrc) {
  asm volatile("global_load_async_to_lds_b128 %0, %1, off"
               :: "v"(lds_dst), "v"(gsrc)
               : "memory");
}
static __device__ inline void async_wait0() {
  asm volatile("s_wait_asynccnt 0" ::: "memory");
}

// ---------------------------------------------------------------------------
// K0: pack theta_w / phi_w into one bf16 Wcat (1024x1024, rows 0..511 = theta,
//     512..1023 = phi) and concatenated fp32 bias (1024).
// ---------------------------------------------------------------------------
__global__ void k_prep(const float* __restrict__ tw, const float* __restrict__ tb,
                       const float* __restrict__ pw, const float* __restrict__ pb,
                       unsigned short* __restrict__ wcat_u,
                       float* __restrict__ bias_cat) {
  __bf16* wcat = (__bf16*)wcat_u;
  int i = blockIdx.x * blockDim.x + threadIdx.x;
  if (i < 1024 * 1024) {
    int r = i >> 10, c = i & 1023;
    float v = (r < 512) ? tw[(size_t)r * 1024 + c]
                        : pw[(size_t)(r - 512) * 1024 + c];
    wcat[i] = (__bf16)v;
  }
  if (i < 1024) bias_cat[i] = (i < 512) ? tb[i] : pb[i - 512];
}

// ---------------------------------------------------------------------------
// K1: per-batch projection GEMM  raw[b] = Wcat(1024x1024) x X_b(1024x192) + bias
//     512 threads = 16 waves; 4 passes over M; each wave: 64x48 accum block.
//     X chunk (32x192) staged fp32->bf16 via VALU; W chunk (256x32) staged
//     via global_load_async_to_lds_b128 (no VGPR bounce, ASYNCcnt tracked).
// ---------------------------------------------------------------------------
__global__ __launch_bounds__(512) void k_proj(
    const float* __restrict__ x, const unsigned short* __restrict__ wcat_u,
    const float* __restrict__ bias_cat, float* __restrict__ raw) {
  __shared__ __bf16 xt[HW_][48];   // X chunk, [n][k], stride 48 halves (96B)
  __shared__ __bf16 wt[256][48];   // W chunk, [m][k], stride 48 halves
  const __bf16* wcat = (const __bf16*)wcat_u;
  const int b = blockIdx.x;
  const int tid = threadIdx.x;
  const int wave = tid >> 5, lane = tid & 31;
  const int lhi = lane >> 4, llo = lane & 15;
  const float* xb = x + (size_t)b * C_ * HW_;
  float* rawb = raw + (size_t)b * C_ * HW_;  // 1024 rows x 192 cols

  for (int p = 0; p < 4; ++p) {
    const int lm0 = (wave >> 2) * 64;           // local M block within W chunk
    const int m0 = p * 256 + lm0;               // global M block
    const int n0 = (wave & 3) * 48;             // N block of 48
    v8f acc[4][3];
#pragma unroll
    for (int a = 0; a < 4; ++a)
#pragma unroll
      for (int c = 0; c < 3; ++c) acc[a][c] = vzero8();

    for (int kc = 0; kc < C_; kc += 32) {
      // stage X[kc:kc+32][0:192] -> xt[n][k] (convert fp32 -> bf16)
      for (int e = tid; e < 32 * HW_; e += 512) {
        int kk = e / HW_;
        int nn = e - kk * HW_;
        const float* src = &xb[(size_t)(kc + kk) * HW_ + nn];
        xt[nn][kk] = (__bf16)(*src);
        if (kc + 32 < C_) __builtin_prefetch(src + 32 * HW_, 0, 1);
      }
      // stage W[p*256 : p*256+256][kc:kc+32] -> wt[m][k] via async-to-LDS
      for (int e = tid; e < 1024; e += 512) {   // 1024 x 16B chunks
        int r = e >> 2, q = e & 3;
        async_copy16((unsigned)(uintptr_t)&wt[r][q * 8],
                     wcat + (size_t)(p * 256 + r) * 1024 + kc + q * 8);
      }
      async_wait0();
      __syncthreads();

      // A fragments (16x32 bf16): lane<16 -> K 0..15; lane>=16 -> K 16..31
      v16bf af[4];
#pragma unroll
      for (int a = 0; a < 4; ++a) {
        const __bf16* ar = &wt[lm0 + a * 16 + llo][lhi * 16];
        v8bf lo = *(const v8bf*)(ar);
        v8bf hi = *(const v8bf*)(ar + 8);
        af[a] = cat16(lo, hi);
      }
      // B fragments (32x16 bf16): lane<16 -> K 0..15 of col n; lane>=16 -> K 16..31
      v16bf bfr[3];
#pragma unroll
      for (int c = 0; c < 3; ++c)
        bfr[c] = *(const v16bf*)&xt[n0 + c * 16 + llo][lhi * 16];

#pragma unroll
      for (int a = 0; a < 4; ++a)
#pragma unroll
        for (int c = 0; c < 3; ++c)
          acc[a][c] = __builtin_amdgcn_wmma_f32_16x16x32_bf16(
              false, af[a], false, bfr[c], (short)0, acc[a][c], false, false);
      __syncthreads();
    }

    // C layout: VGPR r -> row r + 8*(lane>>4), col = lane&15
#pragma unroll
    for (int a = 0; a < 4; ++a)
#pragma unroll
      for (int c = 0; c < 3; ++c)
#pragma unroll
        for (int r = 0; r < 8; ++r) {
          int row = m0 + a * 16 + r + lhi * 8;
          int col = n0 + c * 16 + llo;
          rawb[(size_t)row * HW_ + col] = acc[a][c][r] + bias_cat[row];
        }
  }
}

// ---------------------------------------------------------------------------
// K2: per-(batch,spatial-column) z-score over 512 channels (ddof=1, +1e-4),
//     emit bf16 in K-contiguous layouts: thetaN[b][n][k], phiNT[b][m][k].
// ---------------------------------------------------------------------------
__global__ void k_norm(const float* __restrict__ raw,
                       unsigned short* __restrict__ normb_u) {
  __bf16* normb = (__bf16*)normb_u;
  const int b = blockIdx.x;
  const int col = threadIdx.x;  // blockDim = 192
  for (int proj = 0; proj < 2; ++proj) {
    const float* base = raw + ((size_t)b * 1024 + proj * CI_) * HW_ + col;
    float s = 0.f, ss = 0.f;
    for (int k = 0; k < CI_; ++k) {
      float v = base[(size_t)k * HW_];
      s += v;
      ss += v * v;
    }
    float mean = s * (1.0f / CI_);
    float var = (ss - (float)CI_ * mean * mean) * (1.0f / (CI_ - 1));
    float inv = 1.0f / (sqrtf(fmaxf(var, 0.0f)) + 1e-4f);
    __bf16* out = normb + (((size_t)b * 2 + proj) * HW_ + col) * CI_;
    for (int k = 0; k < CI_; ++k)
      out[k] = (__bf16)((base[(size_t)k * HW_] - mean) * inv);
  }
}

// ---------------------------------------------------------------------------
// K3: per-batch fused: f = thetaN x phiNT / 512 (WMMA, f kept in LDS),
//     t = relu(BN(m1 @ f^T)), mask = sigmoid(m2 @ t), out = x * (1 + mask).
//     512 threads = 16 waves in a 4x4 grid of 48x48 tiles; single K pass.
//     Chunk staging via global_load_async_to_lds_b128.
// ---------------------------------------------------------------------------
__global__ __launch_bounds__(512) void k_corr_mask_apply(
    const unsigned short* __restrict__ normb_u, const float* __restrict__ x,
    const float* __restrict__ m1_w, const float* __restrict__ m1_b,
    const float* __restrict__ bn_g, const float* __restrict__ bn_b,
    const float* __restrict__ bn_m, const float* __restrict__ bn_v,
    const float* __restrict__ m2_w, const float* __restrict__ m2_b,
    float* __restrict__ out) {
  __shared__ float f_lds[HW_][HW_];       // 147456 B
  __shared__ __bf16 ab[2][HW_][48];       //  36864 B (A chunk / B chunk)
  __shared__ float t_lds[64][HW_];        //  49152 B
  __shared__ float msk[HW_];              //    768 B   (total ~229 KB < 320 KB)

  const __bf16* normb = (const __bf16*)normb_u;
  const int b = blockIdx.x;
  const int tid = threadIdx.x;
  const int wave = tid >> 5, lane = tid & 31;
  const int lhi = lane >> 4, llo = lane & 15;
  const __bf16* tN = normb + (size_t)b * 2 * HW_ * CI_;  // theta_x [n][k]
  const __bf16* pN = tN + (size_t)HW_ * CI_;             // phi_x^T [m][k]
  const int n0 = (wave >> 2) * 48, m0 = (wave & 3) * 48;

  v8f acc[3][3];
#pragma unroll
  for (int a = 0; a < 3; ++a)
#pragma unroll
    for (int c = 0; c < 3; ++c) acc[a][c] = vzero8();

  for (int kc = 0; kc < CI_; kc += 32) {
    // stage both 192x32 bf16 chunks: 1536 x 16B async copies (3 per thread)
    for (int e = tid; e < 1536; e += 512) {
      int which = (e >= 768) ? 1 : 0;
      int ee = e - which * 768;
      int n = ee >> 2, q = ee & 3;
      const __bf16* src = (which ? pN : tN) + (size_t)n * CI_ + kc + q * 8;
      async_copy16((unsigned)(uintptr_t)&ab[which][n][q * 8], src);
    }
    async_wait0();
    __syncthreads();

    v16bf af[3], bfr[3];
#pragma unroll
    for (int a = 0; a < 3; ++a) {
      const __bf16* ar = &ab[0][n0 + a * 16 + llo][lhi * 16];
      v8bf lo = *(const v8bf*)(ar);
      v8bf hi = *(const v8bf*)(ar + 8);
      af[a] = cat16(lo, hi);
    }
#pragma unroll
    for (int c = 0; c < 3; ++c)
      bfr[c] = *(const v16bf*)&ab[1][m0 + c * 16 + llo][lhi * 16];

#pragma unroll
    for (int a = 0; a < 3; ++a)
#pragma unroll
      for (int c = 0; c < 3; ++c)
        acc[a][c] = __builtin_amdgcn_wmma_f32_16x16x32_bf16(
            false, af[a], false, bfr[c], (short)0, acc[a][c], false, false);
    __syncthreads();
  }

  // f[n][m] = acc / 512  (reference divides by CI)
#pragma unroll
  for (int a = 0; a < 3; ++a)
#pragma unroll
    for (int c = 0; c < 3; ++c)
#pragma unroll
      for (int r = 0; r < 8; ++r)
        f_lds[n0 + a * 16 + r + lhi * 8][m0 + c * 16 + llo] =
            acc[a][c][r] * (1.0f / CI_);
  __syncthreads();

  // t[j][n] = relu(BN( m1_b[j] + sum_m m1_w[j][m] * f[n][m] ))
  for (int idx = tid; idx < 64 * HW_; idx += 512) {
    int j = idx / HW_;
    int n = idx - j * HW_;
    const float* wr = m1_w + j * HW_;
    float s = m1_b[j];
    for (int m = 0; m < HW_; ++m) s += wr[m] * f_lds[n][m];
    float scale = bn_g[j] * rsqrtf(bn_v[j] + 1e-5f);
    s = (s - bn_m[j]) * scale + bn_b[j];
    t_lds[j][n] = fmaxf(s, 0.0f);
  }
  __syncthreads();

  // mask[n] = 1 + sigmoid(m2_b + sum_j m2_w[j] * t[j][n])
  if (tid < HW_) {
    float s = m2_b[0];
    for (int j = 0; j < 64; ++j) s += m2_w[j] * t_lds[j][tid];
    msk[tid] = 1.0f + 1.0f / (1.0f + __expf(-s));
  }
  __syncthreads();

  // out = x * (1 + mask), coalesced stream
  const float* xb = x + (size_t)b * C_ * HW_;
  float* ob = out + (size_t)b * C_ * HW_;
  for (int e = tid; e < C_ * HW_; e += 512) {
    int n = e % HW_;
    ob[e] = xb[e] * msk[n];
  }
}

// ---------------------------------------------------------------------------
// Launch
// ---------------------------------------------------------------------------
extern "C" void kernel_launch(void* const* d_in, const int* in_sizes, int n_in,
                              void* d_out, int out_size, void* d_ws,
                              size_t ws_size, hipStream_t stream) {
  (void)in_sizes; (void)n_in; (void)out_size; (void)ws_size;
  const float* x   = (const float*)d_in[0];
  const float* tw  = (const float*)d_in[1];
  const float* tb  = (const float*)d_in[2];
  const float* pw  = (const float*)d_in[3];
  const float* pb  = (const float*)d_in[4];
  const float* m1w = (const float*)d_in[5];
  const float* m1b = (const float*)d_in[6];
  const float* bng = (const float*)d_in[7];
  const float* bnb = (const float*)d_in[8];
  const float* bnm = (const float*)d_in[9];
  const float* bnv = (const float*)d_in[10];
  const float* m2w = (const float*)d_in[11];
  const float* m2b = (const float*)d_in[12];
  float* out = (float*)d_out;

  // workspace layout
  char* ws = (char*)d_ws;
  unsigned short* wcat = (unsigned short*)ws;                   // 2 MiB bf16
  float* biasc = (float*)(ws + (size_t)(2u << 20));             // 4 KiB
  float* raw = (float*)(ws + (size_t)(2u << 20) + 4096);        // 100.66 MB f32
  unsigned short* normb =
      (unsigned short*)(ws + (size_t)(2u << 20) + 4096 +
                        (size_t)B_ * C_ * HW_ * sizeof(float)); // 50.33 MB bf16

  k_prep<<<(1024 * 1024 + 255) / 256, 256, 0, stream>>>(tw, tb, pw, pb, wcat,
                                                        biasc);
  k_proj<<<B_, 512, 0, stream>>>(x, wcat, biasc, raw);
  k_norm<<<B_, HW_, 0, stream>>>(raw, normb);
  k_corr_mask_apply<<<B_, 512, 0, stream>>>(normb, x, m1w, m1b, bng, bnb, bnm,
                                            bnv, m2w, m2b, out);
}
